// LSTM_57458072486095
// MI455X (gfx1250) — matrix operand order repaired and checked by reading, not compiled, not run on hardware
//
#include <hip/hip_runtime.h>
#include <hip/hip_bf16.h>
#include <stdint.h>

typedef unsigned short ushort_t;
typedef unsigned int   uint32;
typedef __attribute__((ext_vector_type(16))) __bf16 v16bf;
typedef __attribute__((ext_vector_type(8)))  float  v8f;
typedef __attribute__((ext_vector_type(4)))  uint32 v4u;
typedef __attribute__((ext_vector_type(8)))  uint32 v8u;
typedef __attribute__((ext_vector_type(4)))  float  v4f;

#define BATCH 64
#define SEQ   512
#define ISZ   512
#define HSZ   1024
#define HISZ  1536   // H + I
#define OSZ   512
#define NBLK  16
#define TPB   256

// workspace layout (bytes)
#define WS_WTALL 0
#define WS_WTHY  (12582912)                       // 4096*1536*2
#define WS_HBF   (WS_WTHY + 1048576)              // 512*1024*2
#define WS_CNT   (WS_HBF + 262144)                // 2*64*1024*2
#define WS_XBF   (WS_CNT + 256)
#define WS_NEED_XBF ((size_t)WS_XBF + (size_t)BATCH * SEQ * ISZ * 2)

// ---------------- helpers ----------------

static __device__ __forceinline__ v8f vzero() {
  v8f z;
#pragma unroll
  for (int i = 0; i < 8; ++i) z[i] = 0.0f;
  return z;
}

static __device__ __forceinline__ ushort_t f2bf(float f) {
  uint32 u = __float_as_uint(f);
  u += 0x7FFFu + ((u >> 16) & 1u);   // round-to-nearest-even
  return (ushort_t)(u >> 16);
}

static __device__ __forceinline__ uint32 pk2(float lo, float hi) {
  return (uint32)f2bf(lo) | ((uint32)f2bf(hi) << 16);
}

// A fragment (16x32 bf16): lane half h holds K = h*8 + [0..7] and h*8 + 16 + [0..7]
static __device__ __forceinline__ v16bf ld_fragA_bf16(const ushort_t* p) {
  v4u lo = *(const v4u*)p;
  v4u hi = *(const v4u*)(p + 16);
  v8u c;
  c[0] = lo[0]; c[1] = lo[1]; c[2] = lo[2]; c[3] = lo[3];
  c[4] = hi[0]; c[5] = hi[1]; c[6] = hi[2]; c[7] = hi[3];
  return __builtin_bit_cast(v16bf, c);
}

// B fragment (32x16 bf16): lane half h holds K = h*8 + [0..7] and h*8 + 16 + [0..7]
static __device__ __forceinline__ v16bf ld_fragB_bf16(const ushort_t* p) {
  v4u lo = *(const v4u*)p;
  v4u hi = *(const v4u*)(p + 16);
  v8u c;
  c[0] = lo[0]; c[1] = lo[1]; c[2] = lo[2]; c[3] = lo[3];
  c[4] = hi[0]; c[5] = hi[1]; c[6] = hi[2]; c[7] = hi[3];
  return __builtin_bit_cast(v16bf, c);
}

// A fragment built on-the-fly from fp32 source (fallback when ws too small)
static __device__ __forceinline__ v16bf ld_fragA_f32(const float* p) {
  v4f a = *(const v4f*)(p);
  v4f b = *(const v4f*)(p + 4);
  v4f e = *(const v4f*)(p + 16);
  v4f d = *(const v4f*)(p + 20);
  v8u c;
  c[0] = pk2(a[0], a[1]); c[1] = pk2(a[2], a[3]);
  c[2] = pk2(b[0], b[1]); c[3] = pk2(b[2], b[3]);
  c[4] = pk2(e[0], e[1]); c[5] = pk2(e[2], e[3]);
  c[6] = pk2(d[0], d[1]); c[7] = pk2(d[2], d[3]);
  return __builtin_bit_cast(v16bf, c);
}

static __device__ __forceinline__ v8f wmma_bf16(v16bf a, v16bf b, v8f c) {
  return __builtin_amdgcn_wmma_f32_16x16x32_bf16(false, a, false, b, (short)0, c,
                                                 false, false);
}

static __device__ __forceinline__ float fsig(float x) {
  return __builtin_amdgcn_rcpf(1.0f + __expf(-x));
}
static __device__ __forceinline__ float ftanh_fast(float x) {
  return 2.0f * __builtin_amdgcn_rcpf(1.0f + __expf(-2.0f * x)) - 1.0f;
}

// atomic counter grid barrier (persistent kernel; 16 blocks are co-resident)
static __device__ __forceinline__ void grid_sync(uint32* cnt, uint32 target) {
  __syncthreads();
  if (threadIdx.x == 0) {
    __threadfence();
    __hip_atomic_fetch_add(cnt, 1u, __ATOMIC_RELEASE, __HIP_MEMORY_SCOPE_AGENT);
    while (__hip_atomic_load(cnt, __ATOMIC_ACQUIRE, __HIP_MEMORY_SCOPE_AGENT) < target) {
      __builtin_amdgcn_s_sleep(1);
    }
  }
  __syncthreads();
}

// ---------------- prep kernels ----------------

// W_all^T as bf16: Wt[n][k], n in [0,4096) over gates {f,i,C,o}, k in [0,1536)
__global__ void cvt_gates(const float* __restrict__ Wf, const float* __restrict__ Wi,
                          const float* __restrict__ Wc, const float* __restrict__ Wo,
                          ushort_t* __restrict__ Wt) {
  int idx = blockIdx.x * TPB + threadIdx.x;     // 1536*4096 threads
  int k = idx >> 12;                            // 0..1535
  int n = idx & 4095;                           // 0..4095
  int g = n >> 10, col = n & 1023;
  const float* W = (g == 0) ? Wf : (g == 1) ? Wi : (g == 2) ? Wc : Wo;
  Wt[(size_t)n * HISZ + k] = f2bf(W[(size_t)k * HSZ + col]);
}

// W_hy^T as bf16: Wt[n][k], n in [0,512), k in [0,1024)
__global__ void cvt_hy(const float* __restrict__ Why, ushort_t* __restrict__ Wt) {
  int idx = blockIdx.x * TPB + threadIdx.x;     // 1024*512 threads
  int k = idx >> 9;
  int n = idx & 511;
  Wt[(size_t)n * HSZ + k] = f2bf(Why[(size_t)k * OSZ + n]);
}

// x -> bf16, same [B][S][I] layout (one-shot, HBM-bandwidth bound)
__global__ void cvt_x(const float* __restrict__ x, ushort_t* __restrict__ xb) {
  size_t idx = (size_t)blockIdx.x * TPB + threadIdx.x;   // 64*512*512 threads
  xb[idx] = f2bf(x[idx]);
}

__global__ void init_state(ushort_t* __restrict__ hbf, uint32* __restrict__ cnt) {
  int idx = blockIdx.x * TPB + threadIdx.x;     // 2*64*1024 threads
  hbf[idx] = 0;
  if (idx == 0) *cnt = 0;
}

// ---------------- persistent LSTM kernel ----------------

template <bool USE_XBF>
__global__ __launch_bounds__(TPB, 1) void lstm_persistent(
    const float* __restrict__ x,          // [64][512][512] fp32
    const ushort_t* __restrict__ xbf,     // [64][512][512] bf16 (if USE_XBF)
    const ushort_t* __restrict__ WtAll,   // [4096][1536] bf16 (n-major)
    const ushort_t* __restrict__ WtHy,    // [512][1024]  bf16 (n-major)
    ushort_t* __restrict__ hbf,           // 2 x [64][1024] bf16 (double buffer)
    uint32* __restrict__ cnt,
    const float* __restrict__ b_f, const float* __restrict__ b_i,
    const float* __restrict__ b_c, const float* __restrict__ b_o,
    const float* __restrict__ b_y,
    float* __restrict__ out)              // [64][512][512] fp32
{
  const int lane = threadIdx.x & 31;
  const int wg   = blockIdx.x * (TPB / 32) + (threadIdx.x >> 5);  // 0..127
  const int ln   = lane & 15;
  const int lh   = lane >> 4;
  const int kbA  = lh * 8;    // A-fragment per-lane K base
  const int kbB  = lh * 8;    // B-fragment per-lane K base

  // ---- gates assignment: hidden n-tile + pair of m-tiles ----
  const int gnt  = wg >> 1;            // 0..63 (16 hidden columns each)
  const int gm0  = (wg & 1) << 1;      // m-tiles {gm0, gm0+1}
  const int gcol = gnt * 16 + ln;      // this lane's hidden column

  const float bfv = b_f[gcol];
  const float biv = b_i[gcol];
  const float bcv = b_c[gcol];
  const float bov = b_o[gcol];

  const ushort_t* bpf = WtAll + (size_t)(0 * HSZ + gcol) * HISZ + kbB;
  const ushort_t* bpi = WtAll + (size_t)(1 * HSZ + gcol) * HISZ + kbB;
  const ushort_t* bpc = WtAll + (size_t)(2 * HSZ + gcol) * HISZ + kbB;
  const ushort_t* bpo = WtAll + (size_t)(3 * HSZ + gcol) * HISZ + kbB;

  const size_t aoff0 = (size_t)(gm0 * 16 + ln) * HSZ + kbA;         // h-part A rows
  const size_t aoff1 = aoff0 + (size_t)16 * HSZ;
  const float*    xr0f = x   + (size_t)(gm0 * 16 + ln) * SEQ * ISZ + kbA;
  const float*    xr1f = xr0f + (size_t)16 * SEQ * ISZ;
  const ushort_t* xr0b = xbf + (size_t)(gm0 * 16 + ln) * SEQ * ISZ + kbA;
  const ushort_t* xr1b = xr0b + (size_t)16 * SEQ * ISZ;

  // ---- projection assignment ----
  const int ynt = wg >> 2;             // 0..31
  const int ymt = wg & 3;              // 0..3
  const float byv = b_y[ynt * 16 + ln];
  const size_t yaoff = (size_t)(ymt * 16 + ln) * HSZ + kbA;
  const ushort_t* ybp = WtHy + (size_t)(ynt * 16 + ln) * HSZ + kbB;
  float* outp = out + (size_t)(ymt * 16 + 8 * lh) * SEQ * OSZ + (size_t)(ynt * 16 + ln);

  const size_t HB = (size_t)BATCH * HSZ;

  // cell state lives in registers for the whole scan
  v8f Cst0 = vzero();
  v8f Cst1 = vzero();

  for (int t = 0; t < SEQ; ++t) {
    const ushort_t* hcur = hbf + (size_t)(t & 1) * HB;
    ushort_t*       hnxt = hbf + (size_t)((t + 1) & 1) * HB;

    v8f a00 = vzero(), a01 = vzero(), a02 = vzero(), a03 = vzero();
    v8f a10 = vzero(), a11 = vzero(), a12 = vzero(), a13 = vzero();

    // K over previous hidden state (bf16, double-buffered)
#pragma unroll 4
    for (int kb = 0; kb < HSZ; kb += 32) {
      v16bf A0 = ld_fragA_bf16(hcur + aoff0 + kb);
      v16bf A1 = ld_fragA_bf16(hcur + aoff1 + kb);
      v16bf Bf = ld_fragB_bf16(bpf + kb);
      v16bf Bi = ld_fragB_bf16(bpi + kb);
      v16bf Bc = ld_fragB_bf16(bpc + kb);
      v16bf Bo = ld_fragB_bf16(bpo + kb);
      a00 = wmma_bf16(A0, Bf, a00);  a10 = wmma_bf16(A1, Bf, a10);
      a01 = wmma_bf16(A0, Bi, a01);  a11 = wmma_bf16(A1, Bi, a11);
      a02 = wmma_bf16(A0, Bc, a02);  a12 = wmma_bf16(A1, Bc, a12);
      a03 = wmma_bf16(A0, Bo, a03);  a13 = wmma_bf16(A1, Bo, a13);
    }

    // K over x_t
#pragma unroll 4
    for (int kb = 0; kb < ISZ; kb += 32) {
      v16bf A0, A1;
      if (USE_XBF) {
        A0 = ld_fragA_bf16(xr0b + (size_t)t * ISZ + kb);
        A1 = ld_fragA_bf16(xr1b + (size_t)t * ISZ + kb);
      } else {
        A0 = ld_fragA_f32(xr0f + (size_t)t * ISZ + kb);
        A1 = ld_fragA_f32(xr1f + (size_t)t * ISZ + kb);
      }
      v16bf Bf = ld_fragB_bf16(bpf + HSZ + kb);
      v16bf Bi = ld_fragB_bf16(bpi + HSZ + kb);
      v16bf Bc = ld_fragB_bf16(bpc + HSZ + kb);
      v16bf Bo = ld_fragB_bf16(bpo + HSZ + kb);
      a00 = wmma_bf16(A0, Bf, a00);  a10 = wmma_bf16(A1, Bf, a10);
      a01 = wmma_bf16(A0, Bi, a01);  a11 = wmma_bf16(A1, Bi, a11);
      a02 = wmma_bf16(A0, Bc, a02);  a12 = wmma_bf16(A1, Bc, a12);
      a03 = wmma_bf16(A0, Bo, a03);  a13 = wmma_bf16(A1, Bo, a13);
    }

    // elementwise cell update + h_new (bf16) store, C kept in registers
    {
      ushort_t* hw = hnxt + (size_t)(gm0 * 16 + 8 * lh) * HSZ + gcol;
#pragma unroll
      for (int r = 0; r < 8; ++r) {
        float fv = fsig(a00[r] + bfv);
        float iv = fsig(a01[r] + biv);
        float cv = ftanh_fast(a02[r] + bcv);
        float ov = fsig(a03[r] + bov);
        float Cn = fv * Cst0[r] + iv * cv;
        Cst0[r] = Cn;
        hw[(size_t)r * HSZ] = f2bf(ov * ftanh_fast(Cn));
      }
      hw = hnxt + (size_t)((gm0 + 1) * 16 + 8 * lh) * HSZ + gcol;
#pragma unroll
      for (int r = 0; r < 8; ++r) {
        float fv = fsig(a10[r] + bfv);
        float iv = fsig(a11[r] + biv);
        float cv = ftanh_fast(a12[r] + bcv);
        float ov = fsig(a13[r] + bov);
        float Cn = fv * Cst1[r] + iv * cv;
        Cst1[r] = Cn;
        hw[(size_t)r * HSZ] = f2bf(ov * ftanh_fast(Cn));
      }
    }

    // all h_new visible before anyone reads it (y-GEMM / next step's gates)
    grid_sync(cnt, (uint32)(NBLK * (t + 1)));

    // projection: y_t = h_new @ W_hy + b_y
    v8f ya = vzero();
#pragma unroll 4
    for (int kb = 0; kb < HSZ; kb += 32) {
      v16bf A = ld_fragA_bf16(hnxt + yaoff + kb);
      v16bf B = ld_fragB_bf16(ybp + kb);
      ya = wmma_bf16(A, B, ya);
    }
#pragma unroll
    for (int r = 0; r < 8; ++r) {
      outp[(size_t)r * SEQ * OSZ + (size_t)t * OSZ] = ya[r] + byv;
    }
  }
}

// ---------------- host launcher ----------------

extern "C" void kernel_launch(void* const* d_in, const int* in_sizes, int n_in,
                              void* d_out, int out_size, void* d_ws, size_t ws_size,
                              hipStream_t stream) {
  (void)in_sizes; (void)n_in; (void)out_size;
  // setup_inputs order: x, W_f, W_i, W_o, W_C, b_f, b_i, b_o, b_c, W_hy, b_y
  const float* x   = (const float*)d_in[0];
  const float* Wf  = (const float*)d_in[1];
  const float* Wi  = (const float*)d_in[2];
  const float* Wo  = (const float*)d_in[3];
  const float* Wc  = (const float*)d_in[4];
  const float* bf_ = (const float*)d_in[5];
  const float* bi_ = (const float*)d_in[6];
  const float* bo_ = (const float*)d_in[7];
  const float* bc_ = (const float*)d_in[8];
  const float* Why = (const float*)d_in[9];
  const float* by_ = (const float*)d_in[10];

  char* ws = (char*)d_ws;
  ushort_t* WtAll = (ushort_t*)(ws + WS_WTALL);
  ushort_t* WtHy  = (ushort_t*)(ws + WS_WTHY);
  ushort_t* hbf   = (ushort_t*)(ws + WS_HBF);
  uint32*   cnt   = (uint32*)  (ws + WS_CNT);
  ushort_t* xbf   = (ushort_t*)(ws + WS_XBF);

  const bool use_xbf = (ws_size >= WS_NEED_XBF);

  cvt_gates<<<(4096 * 1536) / TPB, TPB, 0, stream>>>(Wf, Wi, Wc, Wo, WtAll);
  cvt_hy<<<(512 * 1024) / TPB, TPB, 0, stream>>>(Why, WtHy);
  init_state<<<(2 * BATCH * HSZ) / TPB, TPB, 0, stream>>>(hbf, cnt);

  if (use_xbf) {
    cvt_x<<<(BATCH * SEQ * ISZ) / TPB, TPB, 0, stream>>>(x, xbf);
    lstm_persistent<true><<<NBLK, TPB, 0, stream>>>(x, xbf, WtAll, WtHy, hbf, cnt,
                                                    bf_, bi_, bc_, bo_, by_,
                                                    (float*)d_out);
  } else {
    lstm_persistent<false><<<NBLK, TPB, 0, stream>>>(x, xbf, WtAll, WtHy, hbf, cnt,
                                                     bf_, bi_, bc_, bo_, by_,
                                                     (float*)d_out);
  }
}
